// Trainer_65764539236469
// MI455X (gfx1250) — matrix operand order repaired and checked by reading, not compiled, not run on hardware
//
#include <hip/hip_runtime.h>

// ---------------------------------------------------------------------------
// Types for CDNA5 WMMA (wave32): bf16 A/B fragments (v16bf), f32 C/D (v8f)
// ---------------------------------------------------------------------------
typedef __attribute__((ext_vector_type(16))) __bf16         v16bf;
typedef __attribute__((ext_vector_type(8)))  float          v8f;
typedef __attribute__((ext_vector_type(8)))  unsigned short v8u;
typedef __attribute__((ext_vector_type(16))) unsigned short v16u;

#define LDS_LD 264          // 256 + 8 pad -> A-frag lane rows land on distinct bank groups
#define TILE_M 64
#define TILES_PER_NET 128   // 8192 / 64

__device__ __forceinline__ unsigned short f2bf(float f) {
    union { float f; unsigned u; } x; x.f = f;
    unsigned r = x.u + 0x7FFFu + ((x.u >> 16) & 1u);   // round-to-nearest-even
    return (unsigned short)(r >> 16);
}

__device__ __forceinline__ float softplusf(float x) {
    return x > 20.f ? x : log1pf(__expf(fminf(x, 20.f)));
}

// counter-based normal sample: one strong hash + one cheap remix, Box-Muller
__device__ __forceinline__ float gauss(unsigned idx, unsigned layer) {
    unsigned h = idx ^ (layer * 0x9E3779B9u + 0x7F4A7C15u);
    h ^= h >> 16; h *= 0x7FEB352Du; h ^= h >> 15; h *= 0x846CA68Bu; h ^= h >> 16;
    unsigned h2 = (h ^ (h >> 13)) * 0x5BD1E995u;       // cheap second stream
    float u1 = (float)(h  >> 8) * (1.f / 16777216.f) + 1.0e-12f;
    float u2 = (float)(h2 >> 8) * (1.f / 16777216.f);
    return sqrtf(-2.f * __logf(u1)) * __cosf(6.28318530718f * u2);
}

// A fragment (16x32 bf16, M x K) from LDS, row stride LDS_LD.
// ISA layout: lane&15 = row; lane<16 holds K=[0..7]+[16..23], lane>=16 K=[8..15]+[24..31]
__device__ __forceinline__ v16bf load_a(const unsigned short* base, int lane) {
    const unsigned short* p = base + (lane & 15) * LDS_LD + ((lane >> 4) << 3);
    union { struct { v8u lo; v8u hi; } h; v16bf v; } u;
    u.h.lo = *(const v8u*)(p);
    u.h.hi = *(const v8u*)(p + 16);
    return u.v;
}

// Elementwise square of a bf16 fragment (lowers to v_pk_mul_bf16)
__device__ __forceinline__ v16bf sq16(v16bf x) {
    v16bf r;
    #pragma unroll
    for (int i = 0; i < 16; ++i) {
        float f = (float)x[i];
        r[i] = (__bf16)(f * f);
    }
    return r;
}

// B fragment (32x16 bf16, K x N) from column-major weights (k contiguous, col stride 256).
// ISA layout: lane&15 = column; lane<16 holds K=[0..7]+[16..23], lane>=16 K=[8..15]+[24..31]
__device__ __forceinline__ v16bf load_b(const unsigned short* base, int lane) {
    const unsigned short* p = base + (lane & 15) * 256 + ((lane >> 4) << 3);
    union { struct { v8u lo; v8u hi; } h; v16bf v; } w;
    w.h.lo = *(const v8u*)(p);
    w.h.hi = *(const v8u*)(p + 16);
    return w.v;
}

__device__ __forceinline__ v8f wmma_bf16(v16bf a, v16bf b, v8f c) {
    return __builtin_amdgcn_wmma_f32_16x16x32_bf16(false, a, false, b, (short)0, c, false, false);
}

// ---------------------------------------------------------------------------
// Prep: f32 (mu, log_std) -> bf16 column-major (mu_w, std_w^2) + packed biases
// ---------------------------------------------------------------------------
__global__ __launch_bounds__(256)
void prep_weights(const float* __restrict__ mu, const float* __restrict__ ls,
                  unsigned short* __restrict__ Wm, unsigned short* __restrict__ Wv,
                  float2* __restrict__ bmv,
                  int dout, int dopad) {
    const int s = blockIdx.x;
    const int k = threadIdx.x;                     // 0..255 (input-feature index)
    const float* muS = mu + (size_t)s * 257 * dout;
    const float* lsS = ls + (size_t)s * 257 * dout;
    unsigned short* WmS = Wm + (size_t)s * dopad * 256;
    unsigned short* WvS = Wv + (size_t)s * dopad * 256;
    for (int o = 0; o < dopad; ++o) {
        float m = 0.f, v = 0.f;
        if (o < dout) {
            m = muS[(size_t)k * dout + o];
            float sd = softplusf(lsS[(size_t)k * dout + o]);
            v = sd * sd;
        }
        WmS[o * 256 + k] = f2bf(m);
        WvS[o * 256 + k] = f2bf(v);
    }
    // bias row (k == 256 of mu)
    float m = 0.f, v = 0.f;
    if (k < dout) {
        m = muS[(size_t)256 * dout + k];
        float sd = softplusf(lsS[(size_t)256 * dout + k]);
        v = sd * sd;
    }
    bmv[(size_t)s * 256 + k] = make_float2(m, v);
}

// ---------------------------------------------------------------------------
// Fused forward: posenc -> 4x (dual-GEMM + sample + sin) -> final dual-GEMM
// One block = (net s, 64-row tile). 8 waves: each wave = 64 rows x 32 cols
// (4 M-tiles x 2 N-tiles x {mean,var} = 16 accumulators).
// ---------------------------------------------------------------------------
__global__ __launch_bounds__(256)
void fused_siren(const float* __restrict__ X,
                 const unsigned short* __restrict__ WmuH,
                 const unsigned short* __restrict__ WvarH,
                 const unsigned short* __restrict__ Wmu4,
                 const unsigned short* __restrict__ Wvar4,
                 const float2* __restrict__ biasMV,
                 float* __restrict__ out) {
    __shared__ unsigned short sh[TILE_M * LDS_LD];  // activations h (bf16)
    __shared__ float sx[TILE_M * 2];                // staged X tile

    const int s       = blockIdx.x >> 7;
    const int tile    = blockIdx.x & (TILES_PER_NET - 1);
    const int rowBase = tile * TILE_M;
    const int t       = threadIdx.x;
    const int lane    = t & 31;
    const int wave    = t >> 5;        // 0..7 : 32-col slab
    const int colBaseW = wave * 32;

    // stage X tile (64 rows x 2 dims) through LDS
    if (t < TILE_M * 2) sx[t] = X[(size_t)s * 8192 * 2 + (size_t)rowBase * 2 + t];
    __syncthreads();

    // ---------------- positional encoding ----------------
    {
        const int col  = t;               // 0..255 output feature
        const int trig = col >> 7;        // 0 = cos, 1 = sin
        const int f    = col & 127;
        const int d    = f >> 6;          // input dim
        const int j    = f & 63;          // frequency index
        const float c  = 3.14159265358979f * exp2f((float)j * (10.0f / 63.0f));
        for (int r = 0; r < TILE_M; ++r) {
            float ph = c * sx[r * 2 + d];
            float h  = trig ? __sinf(ph) : __cosf(ph);
            sh[r * LDS_LD + col] = f2bf(h);
        }
    }
    __syncthreads();

    const int rOff  = (lane >> 4) * 8;   // C/D tile: rows split by lane half
    const int cLane = lane & 15;

    // ---------------- hidden layers 0..3 ----------------
    #pragma unroll 1
    for (int l = 0; l < 4; ++l) {
        const unsigned short* Wm = WmuH  + ((size_t)l * 64 + s) * 65536;
        const unsigned short* Wv = WvarH + ((size_t)l * 64 + s) * 65536;
        v8f accm[8] = {};   // [mt*2 + nt]
        v8f accv[8] = {};

        #pragma unroll
        for (int k = 0; k < 256; k += 32) {
            v16bf xa[4];
            #pragma unroll
            for (int mt = 0; mt < 4; ++mt)
                xa[mt] = load_a(sh + mt * 16 * LDS_LD + k, lane);

            // mean path: consumes xa
            #pragma unroll
            for (int nt = 0; nt < 2; ++nt) {
                v16bf bmf = load_b(Wm + (size_t)(colBaseW + nt * 16) * 256 + k, lane);
                #pragma unroll
                for (int mt = 0; mt < 4; ++mt)
                    accm[mt * 2 + nt] = wmma_bf16(xa[mt], bmf, accm[mt * 2 + nt]);
            }

            // square in place (v_pk_mul_bf16), then variance path
            #pragma unroll
            for (int mt = 0; mt < 4; ++mt)
                xa[mt] = sq16(xa[mt]);
            #pragma unroll
            for (int nt = 0; nt < 2; ++nt) {
                v16bf bvf = load_b(Wv + (size_t)(colBaseW + nt * 16) * 256 + k, lane);
                #pragma unroll
                for (int mt = 0; mt < 4; ++mt)
                    accv[mt * 2 + nt] = wmma_bf16(xa[mt], bvf, accv[mt * 2 + nt]);
            }
        }
        __syncthreads();   // everyone done reading current activations

        const float2* bmv = biasMV + ((size_t)l * 64 + s) * 256;
        #pragma unroll
        for (int nt = 0; nt < 2; ++nt) {
            const int col = colBaseW + nt * 16 + cLane;
            const float2 b = bmv[col];
            #pragma unroll
            for (int mt = 0; mt < 4; ++mt) {
                #pragma unroll
                for (int r = 0; r < 8; ++r) {
                    const int row = mt * 16 + rOff + r;
                    float m  = accm[mt * 2 + nt][r] + b.x;
                    float sd = sqrtf(accv[mt * 2 + nt][r] + b.y + 1e-14f);
                    unsigned idx = ((unsigned)(s * 8192 + rowBase + row)) * 256u + (unsigned)col;
                    float z = m + sd * gauss(idx, (unsigned)l);
                    sh[row * LDS_LD + col] = f2bf(__sinf(30.0f * z));
                }
            }
        }
        __syncthreads();
    }

    // ---------------- output layer (dout = 3, padded to 16) ----------------
    if (wave < 4) {   // wave-uniform: EXEC stays all-ones for WMMA
        const int mt = wave;              // one 16-row M-tile per wave
        const unsigned short* Wm = Wmu4  + (size_t)s * 16 * 256;
        const unsigned short* Wv = Wvar4 + (size_t)s * 16 * 256;
        v8f am = {};
        v8f av = {};
        #pragma unroll
        for (int k = 0; k < 256; k += 32) {
            v16bf xa  = load_a(sh + mt * 16 * LDS_LD + k, lane);
            v16bf bmf = load_b(Wm + k, lane);
            am = wmma_bf16(xa, bmf, am);
            xa = sq16(xa);
            v16bf bvf = load_b(Wv + k, lane);
            av = wmma_bf16(xa, bvf, av);
        }
        if (cLane < 3) {
            const float2 b = biasMV[((size_t)4 * 64 + s) * 256 + cLane];
            #pragma unroll
            for (int r = 0; r < 8; ++r) {
                const int row = mt * 16 + rOff + r;
                float m  = am[r] + b.x;
                float sd = sqrtf(av[r] + b.y + 1e-14f);
                unsigned idx = ((unsigned)(s * 8192 + rowBase + row)) * 256u + (unsigned)cLane;
                float z = m + sd * gauss(idx, 4u);
                out[((size_t)s * 8192 + rowBase + row) * 3 + cLane] = z;  // last layer: no sin
            }
        }
    }
}

// ---------------------------------------------------------------------------
// Host launcher
// ---------------------------------------------------------------------------
extern "C" void kernel_launch(void* const* d_in, const int* in_sizes, int n_in,
                              void* d_out, int out_size, void* d_ws, size_t ws_size,
                              hipStream_t stream) {
    (void)in_sizes; (void)n_in; (void)out_size; (void)ws_size;
    const float* X = (const float*)d_in[0];
    const float* mu[5]; const float* ls[5];
    for (int i = 0; i < 5; ++i) {
        mu[i] = (const float*)d_in[1 + 2 * i];   // dict order: X, mu0, ls0, mu1, ls1, ...
        ls[i] = (const float*)d_in[2 + 2 * i];
    }

    char* w = (char*)d_ws;
    size_t off = 0;
    auto take = [&](size_t bytes) -> char* {
        char* p = w + off;
        off += (bytes + 255) & ~(size_t)255;
        return p;
    };
    unsigned short* WmuH  = (unsigned short*)take(4ull * 64 * 65536 * 2);  // 33.5 MB
    unsigned short* WvarH = (unsigned short*)take(4ull * 64 * 65536 * 2);  // 33.5 MB
    unsigned short* Wmu4  = (unsigned short*)take(64ull * 16 * 256 * 2);
    unsigned short* Wvar4 = (unsigned short*)take(64ull * 16 * 256 * 2);
    float2* biasMV = (float2*)take(5ull * 64 * 256 * 8);

    for (int l = 0; l < 4; ++l) {
        prep_weights<<<64, 256, 0, stream>>>(mu[l], ls[l],
            WmuH  + (size_t)l * 64 * 65536,
            WvarH + (size_t)l * 64 * 65536,
            biasMV + (size_t)l * 64 * 256,
            256, 256);
    }
    prep_weights<<<64, 256, 0, stream>>>(mu[4], ls[4], Wmu4, Wvar4,
        biasMV + (size_t)4 * 64 * 256, 3, 16);

    fused_siren<<<64 * TILES_PER_NET, 256, 0, stream>>>(
        X, WmuH, WvarH, Wmu4, Wvar4, biasMV, (float*)d_out);
}
